// CrossModalAttentionFusion_58145267253921
// MI455X (gfx1250) — matrix operand order repaired and checked by reading, not compiled, not run on hardware
//
#include <hip/hip_runtime.h>
#include <hip/hip_bf16.h>

typedef __attribute__((ext_vector_type(16))) _Float16 v16h;
typedef __attribute__((ext_vector_type(8)))  _Float16 v8h;
typedef __attribute__((ext_vector_type(4)))  _Float16 v4h;
typedef __attribute__((ext_vector_type(8)))  float    v8f;

#define HW     2304               // 48*48
#define CIN    256
#define HEADS  8
#define DH     32
#define QKV_ELEMS (4 * HEADS * HW * DH)   // 2,359,296 halfs per q/k/v tensor (all modality*batch)

// ---------------------------------------------------------------------------
// WMMA helpers (wave32, 16x16x32 f16 -> f32 accumulate)
// ---------------------------------------------------------------------------
static __device__ __forceinline__ v8f wmma16(v16h a, v16h b, v8f c) {
  return __builtin_amdgcn_wmma_f32_16x16x32_f16(false, a, false, b, (short)0, c, false, false);
}

static __device__ __forceinline__ v8f v8f_zero() {
  v8f z;
#pragma unroll
  for (int i = 0; i < 8; ++i) z[i] = 0.0f;
  return z;
}

// A-matrix fragment (16x32 f16, MxK) from row-major f16, ld in halfs.
// lanes 0-15:  row=lane,    K=0..15  -> e0..15
// lanes 16-31: row=lane-16, K=16..31 -> e0..15
static __device__ __forceinline__ v16h load_afrag(const _Float16* base, int ld) {
  const int lane = threadIdx.x & 31;
  const _Float16* p = base + (lane & 15) * ld + (lane >> 4) * 16;
  v8h lo = *(const v8h*)(p);
  v8h hi = *(const v8h*)(p + 8);
  v16h out;
#pragma unroll
  for (int i = 0; i < 8; ++i) { out[i] = lo[i]; out[i + 8] = hi[i]; }
  return out;
}

// B-matrix fragment (32x16 f16, KxN) from B-transposed rows (N rows, K contiguous).
static __device__ __forceinline__ v16h load_bfragT(const _Float16* bt, int ld) {
  const int lane = threadIdx.x & 31;
  return *(const v16h*)(bt + (lane & 15) * ld + (lane >> 4) * 16);
}

// ---------------------------------------------------------------------------
// Prep 1: f32 -> f16 elementwise (weights keep [o][c] layout == B^T fragment rows)
// ---------------------------------------------------------------------------
__global__ __launch_bounds__(256) void cvt_kernel(const float* __restrict__ src,
                                                  _Float16* __restrict__ dst, int n4) {
  int i = blockIdx.x * blockDim.x + threadIdx.x;
  if (i < n4) {
    float4 v = ((const float4*)src)[i];
    v4h o = { (_Float16)v.x, (_Float16)v.y, (_Float16)v.z, (_Float16)v.w };
    ((v4h*)dst)[i] = o;
  }
}

// ---------------------------------------------------------------------------
// Prep 2: transpose+convert activations: X[mb][c][n] f32 -> XT[mb][n][c] f16
// grid: (72 n-chunks, 8 c-chunks, 4 mb), block 256
// ---------------------------------------------------------------------------
__global__ __launch_bounds__(256) void xt_kernel(const float* __restrict__ ct,
                                                 const float* __restrict__ us,
                                                 _Float16* __restrict__ XTg) {
  __shared__ _Float16 T[32 * 40];
  const int n0 = blockIdx.x * 32;
  const int c0 = blockIdx.y * 32;
  const int mb = blockIdx.z;
  const int m  = mb >> 1;
  const float* X = (m ? us : ct) + (size_t)(mb & 1) * CIN * HW;

  const int tid = threadIdx.x;
  const int cc = tid >> 3;            // channel within chunk
  const int nn = (tid & 7) * 4;       // col within chunk
  float4 v = *(const float4*)(X + (size_t)(c0 + cc) * HW + n0 + nn);
  T[(nn + 0) * 40 + cc] = (_Float16)v.x;
  T[(nn + 1) * 40 + cc] = (_Float16)v.y;
  T[(nn + 2) * 40 + cc] = (_Float16)v.z;
  T[(nn + 3) * 40 + cc] = (_Float16)v.w;
  __syncthreads();

  const int r  = tid >> 3;            // n-row within chunk
  const int cp = (tid & 7) * 4;       // channel quad
  v4h o = { T[r * 40 + cp], T[r * 40 + cp + 1], T[r * 40 + cp + 2], T[r * 40 + cp + 3] };
  *(v4h*)(XTg + ((size_t)mb * HW + n0 + r) * CIN + c0 + cp) = o;
}

// ---------------------------------------------------------------------------
// Kernel 1: QKV GEMM, pure WMMA (no LDS, no conversion).
// D[n, ch] = sum_c XT[n,c] * W[s*256+ch, c]
// grid: (72 n-chunks of 32, 3 = q/k/v, 4 = mb), block 256 = 8 waves (2 n x 4 ch-groups)
// Stores Q,K f16 [mb][h][n][dh]; V transposed f16 [mb][h][dh][n].
// ---------------------------------------------------------------------------
__global__ __launch_bounds__(256) void qkv_kernel(
    const _Float16* __restrict__ XTg,
    const _Float16* __restrict__ w_ct_h, const _Float16* __restrict__ w_us_h,
    _Float16* __restrict__ Qw, _Float16* __restrict__ Kw, _Float16* __restrict__ Vt)
{
  const int n0 = blockIdx.x * 32;
  const int s  = blockIdx.y;               // 0=q 1=k 2=v
  const int mb = blockIdx.z;               // modality*2 + batch

  const _Float16* W = ((mb >> 1) ? w_us_h : w_ct_h) + (size_t)s * 256 * CIN;

  const int tid  = threadIdx.x;
  const int wave = tid >> 5;
  const int lane = tid & 31;
  const int nw = wave & 1;                 // 16-row n half
  const int ow = wave >> 1;                // 64-channel group

  const _Float16* Arow = XTg + ((size_t)mb * HW + n0 + nw * 16) * CIN;

  v8f acc[4];
#pragma unroll
  for (int i = 0; i < 4; ++i) acc[i] = v8f_zero();

#pragma unroll
  for (int cs = 0; cs < CIN; cs += 32) {
    v16h af = load_afrag(Arow + cs, CIN);
#pragma unroll
    for (int osub = 0; osub < 4; ++osub) {
      v16h bf = load_bfragT(W + (size_t)(ow * 64 + osub * 16) * CIN + cs, CIN);
      acc[osub] = wmma16(af, bf, acc[osub]);
    }
  }

  const int hi    = lane >> 4;
  const int nbase = n0 + nw * 16 + hi * 8;   // D rows contiguous across VGPR index
#pragma unroll
  for (int osub = 0; osub < 4; ++osub) {
    const int ch = ow * 64 + osub * 16 + (lane & 15);
    const int h = ch >> 5, d = ch & 31;
    if (s == 2) {
      _Float16* dst = Vt + ((size_t)(mb * HEADS + h) * DH + d) * HW + nbase;
#pragma unroll
      for (int j = 0; j < 8; ++j) dst[j] = (_Float16)acc[osub][j];
    } else {
      _Float16* base = (s == 0 ? Qw : Kw) + (size_t)(mb * HEADS + h) * HW * DH + d;
#pragma unroll
      for (int j = 0; j < 8; ++j) base[(size_t)(nbase + j) * DH] = (_Float16)acc[osub][j];
    }
  }
}

// ---------------------------------------------------------------------------
// Kernel 2: flash-style cross attention; one wave owns a 16-row Q tile.
// Epilogue writes the residual-fused transposed tensor directly:
//   fusedT[b][n][dir*256 + h*32 + d] = XT[mbq][n][h*32+d] + O[n,d]
// grid: (18 groups of 8 q-tiles, 32 = dir*16 + b*8 + h), block 256
// ---------------------------------------------------------------------------
__global__ __launch_bounds__(256) void attn_kernel(
    const _Float16* __restrict__ Qw, const _Float16* __restrict__ Kw,
    const _Float16* __restrict__ Vt, const _Float16* __restrict__ XTg,
    _Float16* __restrict__ fusedT)
{
  __shared__ _Float16 Plds[8 * 16 * 40];   // per-wave 16x32 prob tile (padded)

  const int idx = blockIdx.y;
  const int dir = idx >> 4;
  const int b   = (idx >> 3) & 1;
  const int h   = idx & 7;

  const int tid  = threadIdx.x;
  const int wave = tid >> 5;
  const int lane = tid & 31;
  const int hi = lane >> 4, col = lane & 15;

  const int n0 = (blockIdx.x * 8 + wave) * 16;

  const int mbq  = dir * 2 + b;            // queries from this modality
  const int mbkv = (1 - dir) * 2 + b;      // keys/values from the other one

  const _Float16* Qb = Qw + (size_t)(mbq  * HEADS + h) * HW * DH;
  const _Float16* Kb = Kw + (size_t)(mbkv * HEADS + h) * HW * DH;
  const _Float16* Vb = Vt + (size_t)(mbkv * HEADS + h) * DH * HW;

  const float ssc = 0.17677669529663687f * 1.44269504088896340736f;  // dh^-0.5 * log2(e)

  const v16h qa = load_afrag(Qb + (size_t)n0 * DH, DH);
  const v8f  zf = v8f_zero();

  float mrow[8], lrow[8];
#pragma unroll
  for (int j = 0; j < 8; ++j) { mrow[j] = -1e30f; lrow[j] = 0.0f; }
  v8f o0 = v8f_zero(), o1 = v8f_zero();

  _Float16* pl = &Plds[wave * 16 * 40];

  for (int m0 = 0; m0 < HW; m0 += 32) {
    // S = Q K^T : one WMMA per 16x16 tile (K-dim == dh == 32)
    v16h kb0 = load_bfragT(Kb + (size_t)m0 * DH, DH);
    v16h kb1 = load_bfragT(Kb + (size_t)(m0 + 16) * DH, DH);
    v8f s0 = wmma16(qa, kb0, zf);
    v8f s1 = wmma16(qa, kb1, zf);

    float f[8];
#pragma unroll
    for (int j = 0; j < 8; ++j) {
      float t0 = s0[j] * ssc, t1 = s1[j] * ssc;
      s0[j] = t0; s1[j] = t1;
      float mx = fmaxf(t0, t1);               // row max across the 16-lane half
      mx = fmaxf(mx, __shfl_xor(mx, 1));
      mx = fmaxf(mx, __shfl_xor(mx, 2));
      mx = fmaxf(mx, __shfl_xor(mx, 4));
      mx = fmaxf(mx, __shfl_xor(mx, 8));
      float mn = fmaxf(mrow[j], mx);
      f[j] = exp2f(mrow[j] - mn);
      mrow[j] = mn;
    }
#pragma unroll
    for (int j = 0; j < 8; ++j) {
      float p0 = exp2f(s0[j] - mrow[j]);
      float p1 = exp2f(s1[j] - mrow[j]);
      // D-layout -> A-layout transpose through LDS (in-order DS within a wave)
      pl[(j + 8 * hi) * 40 + col]      = (_Float16)p0;
      pl[(j + 8 * hi) * 40 + 16 + col] = (_Float16)p1;
      float rs = p0 + p1;
      rs += __shfl_xor(rs, 1);
      rs += __shfl_xor(rs, 2);
      rs += __shfl_xor(rs, 4);
      rs += __shfl_xor(rs, 8);
      lrow[j] = lrow[j] * f[j] + rs;
      o0[j] *= f[j];
      o1[j] *= f[j];
    }
    // O += P V  (V stored transposed [d][n] -> contiguous B^T rows)
    v16h pa  = load_afrag(pl, 40);
    v16h vb0 = load_bfragT(Vb + m0, HW);                     // d = 0..15
    v16h vb1 = load_bfragT(Vb + (size_t)16 * HW + m0, HW);   // d = 16..31
    o0 = wmma16(pa, vb0, o0);
    o1 = wmma16(pa, vb1, o1);
  }

#pragma unroll
  for (int j = 0; j < 8; ++j) {
    float inv = 1.0f / lrow[j];
    o0[j] *= inv; o1[j] *= inv;
  }

  // residual add + write fusedT rows (lane: fixed channel, 8 consecutive n rows)
  const _Float16* xr = XTg + ((size_t)mbq * HW + n0 + 8 * hi) * CIN + h * DH + col;
  _Float16* fr = fusedT + ((size_t)b * HW + n0 + 8 * hi) * (2 * CIN) + dir * CIN + h * DH + col;
#pragma unroll
  for (int j = 0; j < 8; ++j) {
    fr[(size_t)j * (2 * CIN)]      = (_Float16)((float)xr[(size_t)j * CIN]      + o0[j]);
    fr[(size_t)j * (2 * CIN) + 16] = (_Float16)((float)xr[(size_t)j * CIN + 16] + o1[j]);
  }
}

// ---------------------------------------------------------------------------
// Kernel 3: output projection, pure WMMA (no LDS, no conversion).
// D[n,o] = sum_{c<512} fusedT[n,c] * w_proj[o,c] + b[o]
// grid: (72 n-chunks of 32, 2 batches), block 256 = 8 waves (2 n x 4 o-groups)
// ---------------------------------------------------------------------------
__global__ __launch_bounds__(256) void proj_kernel(
    const _Float16* __restrict__ fusedT, const _Float16* __restrict__ w_proj_h,
    const float* __restrict__ b_proj, float* __restrict__ out)
{
  const int n0 = blockIdx.x * 32;
  const int b  = blockIdx.y;

  const int tid  = threadIdx.x;
  const int wave = tid >> 5;
  const int lane = tid & 31;
  const int nw = wave & 1, ow = wave >> 1;

  const _Float16* Arow = fusedT + ((size_t)b * HW + n0 + nw * 16) * (2 * CIN);

  v8f acc[4];
#pragma unroll
  for (int i = 0; i < 4; ++i) acc[i] = v8f_zero();

#pragma unroll
  for (int cs = 0; cs < 2 * CIN; cs += 32) {
    v16h af = load_afrag(Arow + cs, 2 * CIN);
#pragma unroll
    for (int osub = 0; osub < 4; ++osub) {
      v16h bf = load_bfragT(w_proj_h + (size_t)(ow * 64 + osub * 16) * (2 * CIN) + cs, 2 * CIN);
      acc[osub] = wmma16(af, bf, acc[osub]);
    }
  }

  const int hi    = lane >> 4;
  const int nbase = n0 + nw * 16 + hi * 8;
#pragma unroll
  for (int osub = 0; osub < 4; ++osub) {
    const int o = ow * 64 + osub * 16 + (lane & 15);
    const float bp = b_proj[o];
    float* dst = out + (size_t)(b * CIN + o) * HW + nbase;
#pragma unroll
    for (int j = 0; j < 8; ++j) dst[j] = acc[osub][j] + bp;
  }
}

// ---------------------------------------------------------------------------
extern "C" void kernel_launch(void* const* d_in, const int* in_sizes, int n_in,
                              void* d_out, int out_size, void* d_ws, size_t ws_size,
                              hipStream_t stream) {
  (void)in_sizes; (void)n_in; (void)out_size; (void)ws_size;

  const float* ct    = (const float*)d_in[0];
  const float* us    = (const float*)d_in[1];
  const float* w_ct  = (const float*)d_in[2];
  const float* w_us  = (const float*)d_in[3];
  const float* wproj = (const float*)d_in[4];
  const float* bproj = (const float*)d_in[5];
  float* out = (float*)d_out;

  // Workspace layout (~24.6 MB), all f16:
  //   XTg    [mb][n][c]      2,359,296 halfs
  //   Qw,Kw  [mb][h][n][dh]  2,359,296 each
  //   Vt     [mb][h][dh][n]  2,359,296
  //   fusedT [b][n][512]     2,359,296
  //   w_ct_h, w_us_h [768][256]; w_proj_h [256][512]
  _Float16* XTg    = (_Float16*)d_ws;
  _Float16* Qw     = XTg + (size_t)4 * HW * CIN;
  _Float16* Kw     = Qw + QKV_ELEMS;
  _Float16* Vt     = Kw + QKV_ELEMS;
  _Float16* fusedT = Vt + QKV_ELEMS;
  _Float16* wcth   = fusedT + (size_t)2 * HW * (2 * CIN);
  _Float16* wush   = wcth + (size_t)768 * CIN;
  _Float16* wprh   = wush + (size_t)768 * CIN;

  // Prep: weight conversion + activation transpose (one-shot, bandwidth-trivial)
  cvt_kernel<<<dim3((768 * CIN / 4 + 255) / 256), 256, 0, stream>>>(w_ct, wcth, 768 * CIN / 4);
  cvt_kernel<<<dim3((768 * CIN / 4 + 255) / 256), 256, 0, stream>>>(w_us, wush, 768 * CIN / 4);
  cvt_kernel<<<dim3((CIN * 2 * CIN / 4 + 255) / 256), 256, 0, stream>>>(wproj, wprh, CIN * 2 * CIN / 4);
  xt_kernel<<<dim3(HW / 32, CIN / 32, 4), 256, 0, stream>>>(ct, us, XTg);

  // Main pipeline
  qkv_kernel<<<dim3(HW / 32, 3, 4), 256, 0, stream>>>(XTg, wcth, wush, Qw, Kw, Vt);
  attn_kernel<<<dim3(HW / 16 / 8, 32), 256, 0, stream>>>(Qw, Kw, Vt, XTg, fusedT);
  proj_kernel<<<dim3(HW / 32, 2), 256, 0, stream>>>(fusedT, wprh, bproj, out);
}
